// TinySwitchGRU_18940805775604
// MI455X (gfx1250) — matrix operand order repaired and checked by reading, not compiled, not run on hardware
//
#include <hip/hip_runtime.h>
#include <hip/hip_bf16.h>

typedef __attribute__((ext_vector_type(16))) _Float16 v16h;
typedef __attribute__((ext_vector_type(8)))  _Float16 v8h;
typedef __attribute__((ext_vector_type(8)))  float    v8f;

#define BB 512
#define TT 512
#define DD 64
#define HH 64
#define OO 2

#define BM 16            // batch rows per block
#define H16_STRIDE 72    // padded f16 row (bank-conflict avoidance, keeps 16B align)
#define H32_STRIDE 68    // padded f32 row

__device__ __forceinline__ v8f wmma16(v16h a, v16h b, v8f c) {
  // D = A(16x32 f16) * B(32x16 f16) + C(16x16 f32)
  return __builtin_amdgcn_wmma_f32_16x16x32_f16(false, a, false, b, (short)0, c, false, false);
}

__device__ __forceinline__ float sigmoidf_(float x) {
  return 1.0f / (1.0f + __expf(-x));
}

__global__ __launch_bounds__(128) void gru_fused(
    const float* __restrict__ x, const int* __restrict__ ctx,
    const unsigned char* __restrict__ mask,
    const float* __restrict__ W_ir, const float* __restrict__ W_iz, const float* __restrict__ W_in,
    const float* __restrict__ b_ir, const float* __restrict__ b_iz, const float* __restrict__ b_in,
    const float* __restrict__ W_hr, const float* __restrict__ W_hz, const float* __restrict__ W_hn,
    const float* __restrict__ b_hr, const float* __restrict__ b_hz, const float* __restrict__ b_hn,
    const float* __restrict__ Wo_w, const float* __restrict__ Wo_b,
    float* __restrict__ out)
{
  // Recurrent weights, pre-swizzled into WMMA-B lane layout (32B/lane per tile):
  //   sWhB flat = ((((g*4 + k)*4 + wave)*2 + c)*32 + lane)*16 + h
  __shared__ _Float16 sWhB[3*4*4*2*32*16];     // 96 KB
  // Input-projection weights, same swizzle (no expert dim):
  //   sWiB flat = (((g*4 + wave)*2 + c)*32 + lane)*16 + h
  __shared__ _Float16 sWiB[3*4*2*32*16];       // 24 KB
  __shared__ _Float16 sX16[16*H16_STRIDE];     // staged x tile (f16, A-operand layout)
  __shared__ _Float16 sH16[16*H16_STRIDE];     // h state as f16 (A operand)
  __shared__ float    sH32[16*H32_STRIDE];     // h state as f32 (output proj)
  __shared__ int      sCtx[16];
  __shared__ float    sM[16];
  __shared__ float    sWo[2*64];
  __shared__ float    sWob[2];

  const int tid  = threadIdx.x;
  const int wave = tid >> 5;
  const int lane = tid & 31;
  const int bm0  = blockIdx.x * BM;

  // ---- setup: stage recurrent weights (transposed) into swizzled LDS, f16 ----
  for (int idx = tid; idx < 3*4*4*2*32*16; idx += 128) {
    int h  = idx & 15;
    int ln = (idx >> 4) & 31;
    int c  = (idx >> 9) & 1;
    int w  = (idx >> 10) & 3;
    int k  = (idx >> 12) & 3;
    int g  = idx >> 14;
    int Kj  = c*32 + (h & 8)*2 + ((ln >= 16) ? 8 : 0) + (h & 7);   // B layout: lanes 0-15 K=0..7,16..23; 16-31 K=8..15,24..31
    int col = w*16 + (ln & 15);
    const float* Wsrc = (g == 0) ? W_hr : ((g == 1) ? W_hz : W_hn);
    sWhB[idx] = (_Float16)Wsrc[(k*64 + col)*64 + Kj];   // B[j=Kj][i=col] = W[k][i][j]
  }
  // ---- setup: input-projection weights, same swizzle ----
  for (int idx = tid; idx < 3*4*2*32*16; idx += 128) {
    int h  = idx & 15;
    int ln = (idx >> 4) & 31;
    int c  = (idx >> 9) & 1;
    int w  = (idx >> 10) & 3;
    int g  = idx >> 12;
    int Kj  = c*32 + (h & 8)*2 + ((ln >= 16) ? 8 : 0) + (h & 7);
    int col = w*16 + (ln & 15);
    const float* Wsrc = (g == 0) ? W_ir : ((g == 1) ? W_iz : W_in);
    sWiB[idx] = (_Float16)Wsrc[col*64 + Kj];            // B[d=Kj][h=col] = W[h][d]
  }
  for (int idx = tid; idx < 16*H16_STRIDE; idx += 128) sH16[idx] = (_Float16)0.0f;
  for (int idx = tid; idx < 16*H32_STRIDE; idx += 128) sH32[idx] = 0.0f;
  for (int idx = tid; idx < 2*64; idx += 128) sWo[idx] = Wo_w[idx];
  if (tid < 2) sWob[tid] = Wo_b[tid];

  // ---- per-lane geometry (CDNA5 wave32 WMMA layouts) ----
  const int colL  = lane & 15;
  const int col   = wave*16 + colL;              // C/D column (output h index)
  const int kOffA = (lane >= 16) ? 8  : 0;       // A-operand K offset / C-D M offset
  const int rowA  = lane & 15;                   // A-operand row (batch row in tile)

  const float biR = b_ir[col], biZ = b_iz[col], biN = b_in[col];
  float bhr[4], bhz[4], bhn[4];
  #pragma unroll
  for (int k = 0; k < 4; ++k) {
    bhr[k] = b_hr[k*64 + col];
    bhz[k] = b_hz[k*64 + col];
    bhn[k] = b_hn[k*64 + col];
  }

  float hreg[8];                                  // live h values (M=v+kOffA, N=col)
  #pragma unroll
  for (int v = 0; v < 8; ++v) hreg[v] = 0.0f;

  // x staging geometry: thread -> (row, 8-col chunk)
  const int xr  = tid >> 3;                       // 0..15
  const int xc0 = (tid & 7) * 8;                  // 0,8,...,56

  __syncthreads();

  for (int t = 0; t < TT; ++t) {
    __syncthreads();                              // sync0: h(t-1)/x(t-1) consumers done
    if (tid < 16) {
      sCtx[tid] = ctx[(size_t)(bm0 + tid)*TT + t];
      sM[tid]   = mask[(size_t)(bm0 + tid)*TT + t] ? 1.0f : 0.0f;
    }

    // ---- cooperative x(t) staging: 1024 floats, each element touched once ----
    {
      const float* px = x + ((size_t)(bm0 + xr)*TT + t)*DD + xc0;
      float4 f0 = *(const float4*)(px);
      float4 f1 = *(const float4*)(px + 4);
      if (t + 1 < TT) __builtin_prefetch(px + DD, 0, 1);   // global_prefetch for x(t+1)
      v8h hv;
      hv[0] = (_Float16)f0.x; hv[1] = (_Float16)f0.y;
      hv[2] = (_Float16)f0.z; hv[3] = (_Float16)f0.w;
      hv[4] = (_Float16)f1.x; hv[5] = (_Float16)f1.y;
      hv[6] = (_Float16)f1.z; hv[7] = (_Float16)f1.w;
      *(v8h*)(sX16 + xr*H16_STRIDE + xc0) = hv;
    }

    // ---- A_h: 16x32 f16 tiles from LDS (aligned 16B reads) ----
    v16h Ah[2];
    {
      const _Float16* hrow = sH16 + rowA*H16_STRIDE;
      #pragma unroll
      for (int c = 0; c < 2; ++c) {
        v8h lo = *(const v8h*)(hrow + c*32 + kOffA);
        v8h hi = *(const v8h*)(hrow + c*32 + 16 + kOffA);
        Ah[c] = __builtin_shufflevector(lo, hi, 0,1,2,3,4,5,6,7,8,9,10,11,12,13,14,15);
      }
    }

    __syncthreads();                              // sync1: x tile + ctx/mask visible

    // ---- A_x: from staged LDS tile ----
    v16h Ax[2];
    {
      const _Float16* xrow = sX16 + rowA*H16_STRIDE;
      #pragma unroll
      for (int c = 0; c < 2; ++c) {
        v8h lo = *(const v8h*)(xrow + c*32 + kOffA);
        v8h hi = *(const v8h*)(xrow + c*32 + 16 + kOffA);
        Ax[c] = __builtin_shufflevector(lo, hi, 0,1,2,3,4,5,6,7,8,9,10,11,12,13,14,15);
      }
    }

    int   ctxv[8];
    float mv[8];
    #pragma unroll
    for (int v = 0; v < 8; ++v) {
      ctxv[v] = sCtx[v + kOffA];
      mv[v]   = sM[v + kOffA];
    }

    // ---- input projections; C seeded with input bias; B tiles from LDS ----
    v8f igR, igZ, igN;
    #pragma unroll
    for (int v = 0; v < 8; ++v) { igR[v] = biR; igZ[v] = biZ; igN[v] = biN; }
    {
      const _Float16* p0 = sWiB + ((size_t)((0*4 + wave)*2)*32 + lane)*16;
      const _Float16* p1 = sWiB + ((size_t)((1*4 + wave)*2)*32 + lane)*16;
      const _Float16* p2 = sWiB + ((size_t)((2*4 + wave)*2)*32 + lane)*16;
      igR = wmma16(Ax[0], *(const v16h*)p0, igR);
      igR = wmma16(Ax[1], *(const v16h*)(p0 + 512), igR);
      igZ = wmma16(Ax[0], *(const v16h*)p1, igZ);
      igZ = wmma16(Ax[1], *(const v16h*)(p1 + 512), igZ);
      igN = wmma16(Ax[0], *(const v16h*)p2, igN);
      igN = wmma16(Ax[1], *(const v16h*)(p2 + 512), igN);
    }

    v8f zero8;
    #pragma unroll
    for (int v = 0; v < 8; ++v) zero8[v] = 0.0f;

    float preR[8], preZ[8], preN[8];

    // ---- gate R: 4 expert accumulators, C = igR ----
    {
      v8f acc[4];
      #pragma unroll
      for (int k = 0; k < 4; ++k) {
        const _Float16* p = sWhB + ((size_t)(((0*4 + k)*4 + wave)*2)*32 + lane)*16;
        v8f a = wmma16(Ah[0], *(const v16h*)p, igR);
        acc[k] = wmma16(Ah[1], *(const v16h*)(p + 512), a);
      }
      #pragma unroll
      for (int v = 0; v < 8; ++v) {
        int kk = ctxv[v];
        float a = acc[0][v];
        a = (kk == 1) ? acc[1][v] : a;
        a = (kk == 2) ? acc[2][v] : a;
        a = (kk == 3) ? acc[3][v] : a;
        float bh = bhr[0];
        bh = (kk == 1) ? bhr[1] : bh;
        bh = (kk == 2) ? bhr[2] : bh;
        bh = (kk == 3) ? bhr[3] : bh;
        preR[v] = a + bh;
      }
    }

    // ---- gate Z: C = igZ ----
    {
      v8f acc[4];
      #pragma unroll
      for (int k = 0; k < 4; ++k) {
        const _Float16* p = sWhB + ((size_t)(((1*4 + k)*4 + wave)*2)*32 + lane)*16;
        v8f a = wmma16(Ah[0], *(const v16h*)p, igZ);
        acc[k] = wmma16(Ah[1], *(const v16h*)(p + 512), a);
      }
      #pragma unroll
      for (int v = 0; v < 8; ++v) {
        int kk = ctxv[v];
        float a = acc[0][v];
        a = (kk == 1) ? acc[1][v] : a;
        a = (kk == 2) ? acc[2][v] : a;
        a = (kk == 3) ? acc[3][v] : a;
        float bh = bhz[0];
        bh = (kk == 1) ? bhz[1] : bh;
        bh = (kk == 2) ? bhz[2] : bh;
        bh = (kk == 3) ? bhz[3] : bh;
        preZ[v] = a + bh;
      }
    }

    // ---- gate N: C = 0 (hn scaled by r before adding in_t) ----
    {
      v8f acc[4];
      #pragma unroll
      for (int k = 0; k < 4; ++k) {
        const _Float16* p = sWhB + ((size_t)(((2*4 + k)*4 + wave)*2)*32 + lane)*16;
        v8f a = wmma16(Ah[0], *(const v16h*)p, zero8);
        acc[k] = wmma16(Ah[1], *(const v16h*)(p + 512), a);
      }
      #pragma unroll
      for (int v = 0; v < 8; ++v) {
        int kk = ctxv[v];
        float a = acc[0][v];
        a = (kk == 1) ? acc[1][v] : a;
        a = (kk == 2) ? acc[2][v] : a;
        a = (kk == 3) ? acc[3][v] : a;
        float bh = bhn[0];
        bh = (kk == 1) ? bhn[1] : bh;
        bh = (kk == 2) ? bhn[2] : bh;
        bh = (kk == 3) ? bhn[3] : bh;
        preN[v] = a + bh;
      }
    }

    // ---- elementwise GRU update + state write-back ----
    #pragma unroll
    for (int v = 0; v < 8; ++v) {
      float r = sigmoidf_(preR[v]);
      float z = sigmoidf_(preZ[v]);
      float n = tanhf(igN[v] + r * preN[v]);
      float hnew = (1.0f - z)*n + z*hreg[v];
      float m = mv[v];
      float h = m*hnew + (1.0f - m)*hreg[v];
      hreg[v] = h;
      int M = v + kOffA;
      sH16[M*H16_STRIDE + col] = (_Float16)h;
      sH32[M*H32_STRIDE + col] = h;
    }

    __syncthreads();                              // sync2: h(t) visible

    // ---- fused output projection (O = 2), wave 0: (row, o) pairs ----
    if (tid < 32) {
      int row = tid >> 1;
      int o   = tid & 1;
      float s = sWob[o];
      const float* wo = sWo + o*64;
      const float* hp = sH32 + row*H32_STRIDE;
      #pragma unroll
      for (int j = 0; j < 64; ++j) s += hp[j] * wo[j];
      out[((size_t)(bm0 + row)*TT + t)*OO + o] = s;
    }
  }
}

extern "C" void kernel_launch(void* const* d_in, const int* in_sizes, int n_in,
                              void* d_out, int out_size, void* d_ws, size_t ws_size,
                              hipStream_t stream) {
  const float*         x    = (const float*)d_in[0];
  const int*           ctx  = (const int*)d_in[1];
  const unsigned char* mask = (const unsigned char*)d_in[2];   // jnp bool_ = 1 byte
  const float* W_ir = (const float*)d_in[3];
  const float* W_iz = (const float*)d_in[4];
  const float* W_in = (const float*)d_in[5];
  const float* b_ir = (const float*)d_in[6];
  const float* b_iz = (const float*)d_in[7];
  const float* b_in = (const float*)d_in[8];
  const float* W_hr = (const float*)d_in[9];
  const float* W_hz = (const float*)d_in[10];
  const float* W_hn = (const float*)d_in[11];
  const float* b_hr = (const float*)d_in[12];
  const float* b_hz = (const float*)d_in[13];
  const float* b_hn = (const float*)d_in[14];
  const float* Wo_w = (const float*)d_in[15];
  const float* Wo_b = (const float*)d_in[16];

  dim3 grid(BB / BM);    // 32 blocks, one per 16-row batch tile (independent recurrences)
  dim3 block(128);       // 4 wave32s: one per 16-wide H tile
  gru_fused<<<grid, block, 0, stream>>>(x, ctx, mask,
                                        W_ir, W_iz, W_in, b_ir, b_iz, b_in,
                                        W_hr, W_hz, W_hn, b_hr, b_hz, b_hn,
                                        Wo_w, Wo_b, (float*)d_out);
}
